// MultiHeadAttention_88845693485106
// MI455X (gfx1250) — compile-verified, matrix operand order checked by reading, not run on hardware
//
#include <hip/hip_runtime.h>

// Problem constants (match reference)
#define BB   2
#define SS   2048
#define DD   2048
#define HH   16
#define HDIM 128

typedef __bf16 bf16x16 __attribute__((ext_vector_type(16)));
typedef float  f32x8   __attribute__((ext_vector_type(8)));
typedef unsigned int u32x4 __attribute__((ext_vector_type(4)));
typedef int  i32x4 __attribute__((ext_vector_type(4)));
typedef int  i32x8 __attribute__((ext_vector_type(8)));

#if defined(__HIP_DEVICE_COMPILE__) && __has_builtin(__builtin_amdgcn_tensor_load_to_lds)
#define USE_TDM 1
#else
#define USE_TDM 0
#endif

__device__ __forceinline__ unsigned short f2bf(float f) {
  unsigned int u = __float_as_uint(f);
  u += 0x7fffu + ((u >> 16) & 1u);   // round-to-nearest-even
  return (unsigned short)(u >> 16);
}

union BfFrag { bf16x16 v; unsigned short s[16]; };

// Build a 16x32 (or 32x16) bf16 WMMA operand for one lane from a row pointer.
// Per ISA: lanes 0-15 hold K chunks {0..7,16..23}, lanes 16-31 hold {8..15,24..31};
// koff = (lane>>4)*8 selects the chunk start.
__device__ __forceinline__ bf16x16 frag_from_row(const unsigned short* base, int koff) {
  BfFrag u;
#pragma unroll
  for (int j = 0; j < 8; ++j) {
    u.s[j]     = base[koff + j];
    u.s[j + 8] = base[koff + 16 + j];
  }
  return u.v;
}

__device__ __forceinline__ f32x8 wmma_bf16(bf16x16 a, bf16x16 b, f32x8 c) {
  return __builtin_amdgcn_wmma_f32_16x16x32_bf16(false, a, false, b, (short)0, c,
                                                 false, false);
}

// One-op butterfly within 16-lane groups via ds_swizzle (group-of-32 xor mode).
template <int XORM>
__device__ __forceinline__ float swz_xor(float x) {
  int i = __builtin_amdgcn_ds_swizzle(__float_as_int(x), (XORM << 10) | 0x1F);
  return __int_as_float(i);
}

#if USE_TDM
// TDM: 2D bf16 tile (rows x cols), global row stride in elements -> LDS with
// +8 bf16 row padding (pad_interval=64 dwords, pad_amount=4 dwords).
__device__ __forceinline__ void tdm_load_2d_bf16_pad(unsigned lds_off,
                                                     const void* gptr,
                                                     int rows, int cols,
                                                     int row_stride_elts) {
  unsigned long long ga = (unsigned long long)(uintptr_t)gptr;
  u32x4 g0;
  g0[0] = 1u;                                            // count=1 (valid D#)
  g0[1] = lds_off;                                       // LDS byte address
  g0[2] = (unsigned)(ga & 0xFFFFFFFFu);                  // global_addr[31:0]
  g0[3] = (unsigned)((ga >> 32) & 0x1FFFFFFu) | (2u << 30); // addr[56:32], type=2
  i32x8 g1;
  // data_size=2B (bits17:16=1), pad_enable(b20), pad_interval=5(64dw, b24:22),
  // pad_amount=3(4dw, b31:25), workgroup_mask=0
  g1[0] = (int)0x07510000u;
  g1[1] = (cols & 0xFFFF) << 16;                         // tensor_dim0[15:0]
  g1[2] = ((cols >> 16) & 0xFFFF) | ((rows & 0xFFFF) << 16); // dim0 hi | dim1 lo
  g1[3] = ((rows >> 16) & 0xFFFF) | ((cols & 0xFFFF) << 16); // dim1 hi | tile_dim0
  g1[4] = rows & 0xFFFF;                                 // tile_dim1 (tile_dim2=0)
  g1[5] = row_stride_elts;                               // tensor_dim0_stride lo
  g1[6] = 0;                                             // stride hi | dim1_stride lo
  g1[7] = 0;
  i32x4 z4 = {0, 0, 0, 0};
#if defined(__clang_major__) && __clang_major__ >= 23
  i32x8 z8 = {0, 0, 0, 0, 0, 0, 0, 0};
  __builtin_amdgcn_tensor_load_to_lds(g0, g1, z4, z4, z8, 0);
#else
  __builtin_amdgcn_tensor_load_to_lds(g0, g1, z4, z4, 0);
#endif
}
#endif

// ---------------------------------------------------------------------------
// C[M,N] = A[M,K] * B[N,K]^T   (torch Linear: y = x @ W.T, W row-major [N,K])
// Block 256 thr = 8 waves; tile 128x128; wave = 32 rows x 64 cols:
// 2 A-frags x 4 B-frags -> 8 WMMA per K-step of 32.
// ---------------------------------------------------------------------------
template <bool A_F32, bool C_F32>
__global__ void __launch_bounds__(256)
gemm_nt(const void* __restrict__ Ap, const float* __restrict__ Bw,
        void* __restrict__ Cp, int M, int N, int K) {
  constexpr int TS = 40;                       // 32 + pad
  __shared__ unsigned short lsA[128 * TS];
  __shared__ unsigned short lsB[128 * TS];
  const int tid  = threadIdx.x;
  const int lane = tid & 31, wid = tid >> 5;
  const int wm = wid >> 1, wn = wid & 1;       // 4 x 2 waves
  const int bm = blockIdx.y * 128, bn = blockIdx.x * 128;
  const int half = lane >> 4, koff = half * 8, ln = lane & 15;
  const int srow = tid >> 1, scol = (tid & 1) * 16;  // staging: 16 elems/thread
  f32x8 acc[2][4];
#pragma unroll
  for (int i = 0; i < 2; ++i)
#pragma unroll
    for (int c = 0; c < 4; ++c) acc[i][c] = (f32x8){};

  for (int k0 = 0; k0 < K; k0 += 32) {
    __syncthreads();
    if (A_F32) {
      const float* src = (const float*)Ap + (size_t)(bm + srow) * K + (k0 + scol);
#pragma unroll
      for (int j = 0; j < 16; ++j) lsA[srow * TS + scol + j] = f2bf(src[j]);
    } else {
      const unsigned short* src =
          (const unsigned short*)Ap + (size_t)(bm + srow) * K + (k0 + scol);
#pragma unroll
      for (int j = 0; j < 16; ++j) lsA[srow * TS + scol + j] = src[j];
    }
    {
      const float* src = Bw + (size_t)(bn + srow) * K + (k0 + scol);
#pragma unroll
      for (int j = 0; j < 16; ++j) lsB[srow * TS + scol + j] = f2bf(src[j]);
    }
    __syncthreads();
    bf16x16 a0 = frag_from_row(&lsA[(wm * 32 + ln) * TS], koff);
    bf16x16 a1 = frag_from_row(&lsA[(wm * 32 + 16 + ln) * TS], koff);
#pragma unroll
    for (int c = 0; c < 4; ++c) {
      bf16x16 b = frag_from_row(&lsB[(wn * 64 + c * 16 + ln) * TS], koff);
      acc[0][c] = wmma_bf16(a0, b, acc[0][c]);
      acc[1][c] = wmma_bf16(a1, b, acc[1][c]);
    }
  }
#pragma unroll
  for (int i = 0; i < 2; ++i)
#pragma unroll
    for (int c = 0; c < 4; ++c) {
      int col = bn + wn * 64 + c * 16 + ln;
#pragma unroll
      for (int r = 0; r < 8; ++r) {
        int row = bm + wm * 32 + i * 16 + half * 8 + r;
        if (C_F32) ((float*)Cp)[(size_t)row * N + col] = acc[i][c][r];
        else ((unsigned short*)Cp)[(size_t)row * N + col] = f2bf(acc[i][c][r]);
      }
    }
}

// ---------------------------------------------------------------------------
// Causal flash attention over bf16 xq/xk/xv ([B,S,H*HD] layout), bf16 xo out.
// Block 128 thr = 4 waves; block owns 64 q rows (16 per wave); 32 keys/tile.
// K tile staged by the Tensor Data Mover (with LDS row padding); V staged
// manually transposed.
// ---------------------------------------------------------------------------
__global__ void __launch_bounds__(128)
flash_fwd(const unsigned short* __restrict__ xq,
          const unsigned short* __restrict__ xk,
          const unsigned short* __restrict__ xv,
          unsigned short* __restrict__ xo) {
  constexpr int KSTR = 136;                    // 128 + 8 (TDM pad_amount)
  constexpr int VSTR = 40;                     // 32 + pad
  constexpr int PSTR = 40;
  __shared__ unsigned short Kl[32 * KSTR];     // K tile, row = key
  __shared__ unsigned short Vt[HDIM * VSTR];   // V tile transposed, row = d
  __shared__ unsigned short Pl[4 * 16 * PSTR]; // per-wave P bounce buffer
  const int tid  = threadIdx.x;
  const int lane = tid & 31, wid = tid >> 5;
  const int half = lane >> 4, koff = half * 8, ln = lane & 15;
  const int q0b = blockIdx.x * 64;
  const int h = blockIdx.y, b = blockIdx.z;
  const int qw = q0b + wid * 16;               // wave's first q row
  const size_t headoff = (size_t)h * HDIM;

  // Preload Q fragments (16 rows x 128 = 4 A-frags), straight from global bf16.
  bf16x16 Qf[4];
  {
    const unsigned short* qbase =
        xq + ((size_t)(b * SS + qw + ln)) * DD + headoff;
#pragma unroll
    for (int f = 0; f < 4; ++f) Qf[f] = frag_from_row(qbase + f * 32, koff);
  }
  f32x8 O[8];
#pragma unroll
  for (int j = 0; j < 8; ++j) O[j] = (f32x8){};
  float mrow[8], lrow[8];
#pragma unroll
  for (int r = 0; r < 8; ++r) { mrow[r] = -1e30f; lrow[r] = 0.0f; }
  const float scale = 0.08838834764831845f;    // 1/sqrt(128)
  const int nkt = (q0b + 64) / 32;             // causal bound for whole block

  for (int kt = 0; kt < nkt; ++kt) {
    const int kb = kt * 32;
    const unsigned short* ksrc0 = xk + ((size_t)(b * SS + kb)) * DD + headoff;
#if USE_TDM
    if (wid == 0)
      tdm_load_2d_bf16_pad((unsigned)(uintptr_t)(void*)Kl, ksrc0, 32, HDIM, DD);
#endif
    {   // V staging (transposed): thread = column c, loop keys
      const int c = tid;
      const unsigned short* vsrc = xv + ((size_t)(b * SS + kb)) * DD + headoff + c;
#pragma unroll 4
      for (int r = 0; r < 32; ++r)
        Vt[c * VSTR + r] = vsrc[(size_t)r * DD];   // transpose on the fly
#if !USE_TDM
      const unsigned short* ksrc = ksrc0 + c;
#pragma unroll 4
      for (int r = 0; r < 32; ++r)
        Kl[r * KSTR + c] = ksrc[(size_t)r * DD];
#endif
    }
#if USE_TDM
    if (wid == 0) __builtin_amdgcn_s_wait_tensorcnt(0);
#endif
    __syncthreads();

    // scores: 16 q x 32 keys = 2 C-frags, 4 WMMA each over HD=128
    f32x8 sc[2]; sc[0] = (f32x8){}; sc[1] = (f32x8){};
#pragma unroll
    for (int g = 0; g < 2; ++g)
#pragma unroll
      for (int f = 0; f < 4; ++f) {
        bf16x16 bk = frag_from_row(&Kl[(g * 16 + ln) * KSTR + f * 32], koff);
        sc[g] = wmma_bf16(Qf[f], bk, sc[g]);
      }

    // scale + causal mask in C-fragment layout (lane%16 = key col, r = row)
#pragma unroll
    for (int g = 0; g < 2; ++g) {
      int ncol = kb + g * 16 + ln;
#pragma unroll
      for (int r = 0; r < 8; ++r) {
        int mr = qw + half * 8 + r;
        float s = sc[g][r] * scale;
        sc[g][r] = (ncol > mr) ? -1e30f : s;
      }
    }

    // online softmax: row stats live in one 16-lane half -> swizzle-xor reduce
    float alpha[8];
#pragma unroll
    for (int r = 0; r < 8; ++r) {
      float rm = fmaxf(sc[0][r], sc[1][r]);
      rm = fmaxf(rm, swz_xor<1>(rm));
      rm = fmaxf(rm, swz_xor<2>(rm));
      rm = fmaxf(rm, swz_xor<4>(rm));
      rm = fmaxf(rm, swz_xor<8>(rm));
      float mnew = fmaxf(mrow[r], rm);
      alpha[r] = __expf(mrow[r] - mnew);
      mrow[r] = mnew;
      float p0 = __expf(sc[0][r] - mnew);
      float p1 = __expf(sc[1][r] - mnew);
      sc[0][r] = p0; sc[1][r] = p1;
      float rs = p0 + p1;
      rs += swz_xor<1>(rs);
      rs += swz_xor<2>(rs);
      rs += swz_xor<4>(rs);
      rs += swz_xor<8>(rs);
      lrow[r] = lrow[r] * alpha[r] + rs;
    }
#pragma unroll
    for (int j = 0; j < 8; ++j)
#pragma unroll
      for (int r = 0; r < 8; ++r) O[j][r] *= alpha[r];

    // P: C-layout regs -> wave-private LDS -> reload in A-layout
    unsigned short* Pw = &Pl[wid * 16 * PSTR];
#pragma unroll
    for (int g = 0; g < 2; ++g)
#pragma unroll
      for (int r = 0; r < 8; ++r)
        Pw[(half * 8 + r) * PSTR + g * 16 + ln] = f2bf(sc[g][r]);
    asm volatile("s_wait_dscnt 0" ::: "memory");  // wave-local LDS RAW fence
    bf16x16 Pa = frag_from_row(&Pw[ln * PSTR], koff);

    // O += P[16x32] x V[32x128]: 8 WMMA, B-frags contiguous from Vt rows
#pragma unroll
    for (int j = 0; j < 8; ++j) {
      bf16x16 bv = frag_from_row(&Vt[(j * 16 + ln) * VSTR], koff);
      O[j] = wmma_bf16(Pa, bv, O[j]);
    }
    __syncthreads();
  }

  // normalize + store bf16
  float invl[8];
#pragma unroll
  for (int r = 0; r < 8; ++r) invl[r] = 1.0f / lrow[r];
#pragma unroll
  for (int j = 0; j < 8; ++j) {
    size_t col = headoff + j * 16 + ln;
#pragma unroll
    for (int r = 0; r < 8; ++r) {
      int row = qw + half * 8 + r;
      xo[((size_t)(b * SS + row)) * DD + col] = f2bf(O[j][r] * invl[r]);
    }
  }
}

extern "C" void kernel_launch(void* const* d_in, const int* in_sizes, int n_in,
                              void* d_out, int out_size, void* d_ws, size_t ws_size,
                              hipStream_t stream) {
  (void)in_sizes; (void)n_in; (void)out_size; (void)ws_size;
  const float* q  = (const float*)d_in[0];
  const float* k  = (const float*)d_in[1];
  const float* v  = (const float*)d_in[2];
  const float* wq = (const float*)d_in[3];
  const float* wk = (const float*)d_in[4];
  const float* wv = (const float*)d_in[5];
  const float* wo = (const float*)d_in[6];

  const size_t BSD = (size_t)BB * SS * DD;     // 8M elements
  unsigned short* xq = (unsigned short*)d_ws;  // 4 bf16 buffers = 64 MB total
  unsigned short* xk = xq + BSD;
  unsigned short* xv = xk + BSD;
  unsigned short* xo = xv + BSD;

  const int M = BB * SS;                       // 4096
  dim3 ggrid(DD / 128, M / 128);               // (16, 32)
  dim3 gblk(256);

  gemm_nt<true, false><<<ggrid, gblk, 0, stream>>>(q, wq, xq, M, DD, DD);
  gemm_nt<true, false><<<ggrid, gblk, 0, stream>>>(k, wk, xk, M, DD, DD);
  gemm_nt<true, false><<<ggrid, gblk, 0, stream>>>(v, wv, xv, M, DD, DD);

  dim3 fgrid(SS / 64, HH, BB);                 // (32, 16, 2)
  flash_fwd<<<fgrid, dim3(128), 0, stream>>>(xq, xk, xv, xo);

  gemm_nt<false, true><<<ggrid, gblk, 0, stream>>>(xo, wo, (float*)d_out, M, DD, DD);
}
